// Generator_43482248905036
// MI455X (gfx1250) — matrix operand order repaired and checked by reading, not compiled, not run on hardware
//
#include <hip/hip_runtime.h>
#include <hip/hip_bf16.h>

// Problem constants (match reference)
#define B_ 256
#define T_ 256
#define D_ 128
#define H_ 512

typedef __attribute__((ext_vector_type(16))) __bf16 v16bf;
typedef __attribute__((ext_vector_type(8)))  __bf16 v8bf;
typedef __attribute__((ext_vector_type(8)))  float  v8f;

// ---------------------------------------------------------------------------
// CDNA5 WMMA 16x16x32 bf16 fragment layouts (ISA 7.12.2):
// A (16x32 MxK): lanes 0-15 : M=lane,    K={k0..k0+7, k0+16..k0+23}
//                lanes 16-31: M=lane-16, K={k0+8..k0+15, k0+24..k0+31}
// B (32x16 KxN): lanes 0-15 : N=lane,    K={k0..k0+7, k0+16..k0+23}
//                lanes 16-31: N=lane-16, K={k0+8..k0+15, k0+24..k0+31}
// C/D (16x16):   vgpr r: lanes 0-15 -> M=r, lanes 16-31 -> M=r+8; N=lane&15
//
// Weights and x are pre-swizzled (one-time pass) into per-(tile,lane)
// contiguous v16bf fragments; per step the weight fragments for one N-tile
// are staged into LDS once and shared by 8 waves (8 batch tiles).
// ---------------------------------------------------------------------------

__device__ __forceinline__ v8f wmma_bf16(v16bf a, v16bf b, v8f c) {
  return __builtin_amdgcn_wmma_f32_16x16x32_bf16(
      /*neg_a=*/false, a, /*neg_b=*/false, b,
      /*c_mod=*/(short)0, c, /*reuse_a=*/false, /*reuse_b=*/false);
}

__device__ __forceinline__ float sigmoid_f(float v) {
  return 1.0f / (1.0f + __expf(-v));
}

// Pre-swizzled fragment fetch: frags laid out [ntile][ktile][lane][16]
__device__ __forceinline__ v16bf load_frag(const __bf16* __restrict__ base,
                                           int ntile, int ktile, int nKt,
                                           int lane) {
  return ((const v16bf*)base)[((size_t)ntile * nKt + ktile) * 32 + lane];
}

// A-fragment of hidden state from row-major bf16 [B,H]
__device__ __forceinline__ v16bf load_a_h(const __bf16* __restrict__ hbf,
                                          int m, int k0, int half) {
  const __bf16* p = hbf + (size_t)m * H_ + k0 + half * 8;  // 16B aligned
  v8bf lo = *(const v8bf*)p;
  v8bf hi = *(const v8bf*)(p + 16);
  v16bf a;
#pragma unroll
  for (int i = 0; i < 8; ++i) { a[i] = lo[i]; a[8 + i] = hi[i]; }
  return a;
}

// ---------------------------------------------------------------------------
// One-time pre-pass: swizzle a row-major fp32 weight [Ntot x K] into bf16
// B-fragments. grid = (Ntot/16, K/32), block = 32.
// ---------------------------------------------------------------------------
__global__ void swizzle_w_kernel(const float* __restrict__ W,
                                 __bf16* __restrict__ out, int K) {
  const int lane = threadIdx.x;
  const int nt = blockIdx.x, kt = blockIdx.y, nKt = gridDim.y;
  const float* p =
      W + (size_t)(nt * 16 + (lane & 15)) * K + kt * 32 + (lane >> 4) * 8;
  v16bf v;
#pragma unroll
  for (int i = 0; i < 8; ++i) { v[i] = (__bf16)p[i]; v[8 + i] = (__bf16)p[16 + i]; }
  ((v16bf*)out)[((size_t)nt * nKt + kt) * 32 + lane] = v;
}

// ---------------------------------------------------------------------------
// One-time pre-pass: swizzle x [B,T,D] fp32 into bf16 A-fragments laid out
// [T][B/16][D/32][lane][16]. grid = (T, B/16, D/32), block = 32.
// ---------------------------------------------------------------------------
__global__ void swizzle_x_kernel(const float* __restrict__ x,
                                 __bf16* __restrict__ out) {
  const int lane = threadIdx.x;
  const int t = blockIdx.x, bt = blockIdx.y, kt = blockIdx.z;
  const int m = bt * 16 + (lane & 15);
  const float* p = x + ((size_t)m * T_ + t) * D_ + kt * 32 + (lane >> 4) * 8;
  v16bf v;
#pragma unroll
  for (int i = 0; i < 8; ++i) {
    v[i]     = (__bf16)p[i];
    v[8 + i] = (__bf16)p[16 + i];
  }
  ((v16bf*)out)[(((size_t)t * (B_ / 16) + bt) * (D_ / 32) + kt) * 32 + lane] = v;
}

// ---------------------------------------------------------------------------
// One GRU layer step.  Block = 8 waves sharing one hidden N-tile; wave w
// handles batch tile bt = blockIdx.x*8 + w.  All 6 weight fragment chunks
// for this N-tile (r/z/n x w_ih,w_hh) are staged into LDS once, then each
// wave runs the fused 60-WMMA tile:  gi + gh + gate nonlinearity.
// grid = (B/16/8, H/16) = (2, 32), block = 256.
// ---------------------------------------------------------------------------
#define XKT (D_ / 32)              /* 4  */
#define HKT (H_ / 32)              /* 16 */
#define WIH_FRAGS (3 * XKT * 32)   /* 384  v16bf = 12 KB */
#define WHH_FRAGS (3 * HKT * 32)   /* 1536 v16bf = 48 KB */

__global__ void __launch_bounds__(256)
gru_step_kernel(const __bf16* __restrict__ xf,    // swizzled x fragments
                const float*  __restrict__ h_f,   // [B,H] f32
                const __bf16* __restrict__ h_bf,  // [B,H] bf16
                float*  __restrict__ ho_f,        // [B,H] f32
                __bf16* __restrict__ ho_bf,       // [B,H] bf16
                const __bf16* __restrict__ wih,   // frags 3H x D
                const __bf16* __restrict__ whh,   // frags 3H x H
                const float* __restrict__ b_ih,   // [3H]
                const float* __restrict__ b_hh,   // [3H]
                int t) {
  const int tid  = threadIdx.x;
  const int lane = tid & 31;
  const int wave = tid >> 5;
  const int half = lane >> 4;
  const int nt   = blockIdx.y;              // hidden tile (of H/16)
  const int bt   = blockIdx.x * 8 + wave;   // batch tile  (of B/16)
  const int b0   = bt * 16;
  const int n0   = nt * 16;

  __shared__ v16bf lds_wih[WIH_FRAGS];   // [gate][kt][lane]
  __shared__ v16bf lds_whh[WHH_FRAGS];   // [gate][kt][lane]

  // ---- cooperative staging: weight fragments for this N-tile -> LDS ----
  const v16bf* wihf = (const v16bf*)wih;
  const v16bf* whhf = (const v16bf*)whh;
#pragma unroll
  for (int i = tid; i < WIH_FRAGS; i += 256) {
    const int g = i / (XKT * 32);
    const int r = i - g * (XKT * 32);
    lds_wih[i] = wihf[(size_t)(nt + g * (H_ / 16)) * (XKT * 32) + r];
  }
#pragma unroll
  for (int i = tid; i < WHH_FRAGS; i += 256) {
    const int g = i / (HKT * 32);
    const int r = i - g * (HKT * 32);
    lds_whh[i] = whhf[(size_t)(nt + g * (H_ / 16)) * (HKT * 32) + r];
  }
  __syncthreads();

  v8f acc_r = {}, acc_z = {}, acc_ni = {}, acc_nh = {};

  // ---- input contribution: x_t @ w_ih^T, K over D ----
  const v16bf* xfr =
      (const v16bf*)xf + ((size_t)t * (B_ / 16) + bt) * XKT * 32;
#pragma unroll 2
  for (int kt = 0; kt < XKT; ++kt) {
    v16bf a  = xfr[kt * 32 + lane];
    v16bf br = lds_wih[0 * (XKT * 32) + kt * 32 + lane];
    v16bf bz = lds_wih[1 * (XKT * 32) + kt * 32 + lane];
    v16bf bn = lds_wih[2 * (XKT * 32) + kt * 32 + lane];
    acc_r  = wmma_bf16(a, br, acc_r);
    acc_z  = wmma_bf16(a, bz, acc_z);
    acc_ni = wmma_bf16(a, bn, acc_ni);
  }

  // ---- recurrent contribution: h @ w_hh^T, K over H ----
  const int mrow = b0 + (lane & 15);
#pragma unroll 2
  for (int kt = 0; kt < HKT; ++kt) {
    v16bf a  = load_a_h(h_bf, mrow, kt * 32, half);
    v16bf br = lds_whh[0 * (HKT * 32) + kt * 32 + lane];
    v16bf bz = lds_whh[1 * (HKT * 32) + kt * 32 + lane];
    v16bf bn = lds_whh[2 * (HKT * 32) + kt * 32 + lane];
    acc_r  = wmma_bf16(a, br, acc_r);
    acc_z  = wmma_bf16(a, bz, acc_z);
    acc_nh = wmma_bf16(a, bn, acc_nh);
  }

  // ---- gate math + state update ----
  const int n = lane & 15;
  const float br_ = b_ih[n0 + n] + b_hh[n0 + n];
  const float bz_ = b_ih[H_ + n0 + n] + b_hh[H_ + n0 + n];
  const float bni = b_ih[2 * H_ + n0 + n];
  const float bnh = b_hh[2 * H_ + n0 + n];

#pragma unroll
  for (int r = 0; r < 8; ++r) {
    const int m = b0 + half * 8 + r;
    const float rg = sigmoid_f(acc_r[r] + br_);
    const float zg = sigmoid_f(acc_z[r] + bz_);
    const float ng = tanhf(acc_ni[r] + bni + rg * (acc_nh[r] + bnh));
    const float ho = h_f[(size_t)m * H_ + n0 + n];
    const float hv = (1.0f - zg) * ng + zg * ho;
    ho_f[(size_t)m * H_ + n0 + n]  = hv;
    ho_bf[(size_t)m * H_ + n0 + n] = (__bf16)hv;
  }
}

// ---------------------------------------------------------------------------
// Projection + imputation/completion for timestep t (t < T-1):
//   imp  = h @ w_proj^T + b_proj ;  comp = m_t*x_t + (1-m_t)*imp
// grid = (B/16, D/16), block = 32.
// ---------------------------------------------------------------------------
__global__ void proj_kernel(const float* __restrict__ x,       // [B,T,D]
                            const float* __restrict__ masks,   // [B,T]
                            const __bf16* __restrict__ h_bf,   // [B,H]
                            const __bf16* __restrict__ wproj,  // frags D x H
                            const float* __restrict__ b_proj,  // [D]
                            float* __restrict__ completed,     // [B,T-1,D]
                            float* __restrict__ imputation,    // [B,T-1,D]
                            int t) {
  const int lane = threadIdx.x;
  const int half = lane >> 4;
  const int b0   = blockIdx.x * 16;
  const int nt   = blockIdx.y;
  const int n0   = nt * 16;
  const int mrow = b0 + (lane & 15);

  v8f acc = {};
#pragma unroll 4
  for (int kt = 0; kt < HKT; ++kt) {
    v16bf a = load_a_h(h_bf, mrow, kt * 32, half);
    v16bf b = load_frag(wproj, nt, kt, HKT, lane);
    acc = wmma_bf16(a, b, acc);
  }

  const int n = lane & 15;
  const float bias = b_proj[n0 + n];
#pragma unroll
  for (int r = 0; r < 8; ++r) {
    const int m = b0 + half * 8 + r;
    const float imp = acc[r] + bias;
    const float xv  = x[((size_t)m * T_ + t) * D_ + n0 + n];
    const float mv  = masks[(size_t)m * T_ + t];
    const size_t o  = ((size_t)m * (T_ - 1) + t) * D_ + n0 + n;
    imputation[o] = imp;
    completed[o]  = mv * xv + (1.0f - mv) * imp;
  }
}

// ---------------------------------------------------------------------------
// kernel_launch
// ---------------------------------------------------------------------------
extern "C" void kernel_launch(void* const* d_in, const int* in_sizes, int n_in,
                              void* d_out, int out_size, void* d_ws, size_t ws_size,
                              hipStream_t stream) {
  const float* x      = (const float*)d_in[0];  // [B,T,D]
  const float* masks  = (const float*)d_in[1];  // [B,T]
  const float* w_ih   = (const float*)d_in[2];  // [L,3H,D]
  const float* w_hh   = (const float*)d_in[3];  // [L,3H,H]
  const float* b_ih   = (const float*)d_in[4];  // [L,3H]
  const float* b_hh   = (const float*)d_in[5];  // [L,3H]
  const float* w_proj = (const float*)d_in[6];  // [D,H]
  const float* b_proj = (const float*)d_in[7];  // [D]

  float* out        = (float*)d_out;
  float* h_final    = out;                                    // [B,H]
  float* completed  = out + (size_t)B_ * H_;                  // [B,T-1,D]
  float* imputation = completed + (size_t)B_ * (T_ - 1) * D_; // [B,T-1,D]

  // ---- workspace carve-up ----
  char* ws = (char*)d_ws;
  float* h_a = (float*)ws;                 ws += (size_t)B_ * H_ * 4;  // 512K
  float* h_b = (float*)ws;                 ws += (size_t)B_ * H_ * 4;  // 512K
  __bf16* hbf_a = (__bf16*)ws;             ws += (size_t)B_ * H_ * 2;  // 256K
  __bf16* hbf_b = (__bf16*)ws;             ws += (size_t)B_ * H_ * 2;  // 256K
  __bf16* wih_sw[2], *whh_sw[2];
  for (int l = 0; l < 2; ++l) { wih_sw[l] = (__bf16*)ws; ws += (size_t)3 * H_ * D_ * 2; }
  for (int l = 0; l < 2; ++l) { whh_sw[l] = (__bf16*)ws; ws += (size_t)3 * H_ * H_ * 2; }
  __bf16* wproj_sw = (__bf16*)ws;          ws += (size_t)D_ * H_ * 2;
  __bf16* x_sw = (__bf16*)ws;              ws += (size_t)B_ * T_ * D_ * 2;

  // ---- one-time pre-passes: convert + swizzle weights and x to bf16 ----
  for (int l = 0; l < 2; ++l) {
    swizzle_w_kernel<<<dim3(3 * H_ / 16, D_ / 32), 32, 0, stream>>>(
        w_ih + (size_t)l * 3 * H_ * D_, wih_sw[l], D_);
    swizzle_w_kernel<<<dim3(3 * H_ / 16, H_ / 32), 32, 0, stream>>>(
        w_hh + (size_t)l * 3 * H_ * H_, whh_sw[l], H_);
  }
  swizzle_w_kernel<<<dim3(D_ / 16, H_ / 32), 32, 0, stream>>>(w_proj, wproj_sw, H_);
  swizzle_x_kernel<<<dim3(T_, B_ / 16, D_ / 32), 32, 0, stream>>>(x, x_sw);

  hipMemsetAsync(h_a,   0, (size_t)B_ * H_ * sizeof(float),  stream);
  hipMemsetAsync(hbf_a, 0, (size_t)B_ * H_ * sizeof(__bf16), stream);

  dim3 gate_grid(B_ / 16 / 8, H_ / 16);  // (2, 32), 8 waves/block
  dim3 proj_grid(B_ / 16, D_ / 16);      // (16, 8)

  for (int t = 0; t < T_; ++t) {
    gru_step_kernel<<<gate_grid, 256, 0, stream>>>(
        x_sw, h_a, hbf_a, h_b, hbf_b, wih_sw[0], whh_sw[0], b_ih, b_hh, t);
    gru_step_kernel<<<gate_grid, 256, 0, stream>>>(
        x_sw, h_b, hbf_b, h_a, hbf_a, wih_sw[1], whh_sw[1],
        b_ih + 3 * H_, b_hh + 3 * H_, t);
    if (t < T_ - 1) {
      proj_kernel<<<proj_grid, 32, 0, stream>>>(
          x, masks, hbf_a, wproj_sw, b_proj, completed, imputation, t);
    }
  }

  hipMemcpyAsync(h_final, h_a, (size_t)B_ * H_ * sizeof(float),
                 hipMemcpyDeviceToDevice, stream);
}
